// PaiNNMixing_72344429134052
// MI455X (gfx1250) — compile-verified, matrix operand order checked by reading, not run on hardware
//
#include <hip/hip_runtime.h>
#include <hip/hip_bf16.h>

typedef __attribute__((ext_vector_type(16))) _Float16 v16h;
typedef __attribute__((ext_vector_type(8)))  _Float16 v8h;
typedef __attribute__((ext_vector_type(8)))  float    v8f;

union Frag {
  v16h h;
  uint4 q[2];
  unsigned int u[8];
};

__device__ __forceinline__ v8f wmma_f16(v16h a, v16h b, v8f c) {
  // D = A(16x32 f16) * B(32x16 f16) + C(16x16 f32)
  return __builtin_amdgcn_wmma_f32_16x16x32_f16(false, a, false, b, (short)0, c,
                                                false, false);
}

// B-fragments are pre-swizzled in workspace: frag[fragIdx][lane][0..15] halves.
__device__ __forceinline__ v16h load_bfrag(const _Float16* base, int fragIdx, int lane) {
  const uint4* p = reinterpret_cast<const uint4*>(base + ((size_t)fragIdx * 32 + lane) * 16);
  Frag f;
  f.q[0] = p[0];
  f.q[1] = p[1];
  return f.h;
}

// A-fragment gather from a row-major f16 LDS tile (row = lane&15).
// 16-bit A layout (ISA 7.12.2): lanes0-15 K={0..7,16..23}, lanes16-31 K={8..15,24..31},
// packed as consecutive K-pairs per dword -> 8 ds dword loads (merged to b128 by clauses).
__device__ __forceinline__ v16h load_afrag(const _Float16* row, int kt, int hi4) {
  const unsigned int* r = reinterpret_cast<const unsigned int*>(row);
  Frag f;
  int b = kt * 16 + hi4;
#pragma unroll
  for (int v = 0; v < 8; ++v) f.u[v] = r[b + (v < 4 ? v : v + 4)];
  return f.h;
}

// ---------------------------------------------------------------------------
// Prep: convert f32 weights -> f16 fragments in the exact B-operand VGPR order.
// frags layout: [0,64)   W_mu  (kt=4, nt=16)
//               [64,128) W1    (kt=8, nt=8)
//               [128,224) W2   (kt=4, nt=24)
// ---------------------------------------------------------------------------
__global__ void painn_prep(const float* __restrict__ Wmu, const float* __restrict__ W1,
                           const float* __restrict__ W2, _Float16* __restrict__ frags) {
  int fi = blockIdx.x;        // 0..223
  int lane = threadIdx.x;     // 0..31
  const float* src;
  int kt, nt, stride, fragLocal;
  _Float16* dst;
  if (fi < 64) {
    src = Wmu; stride = 256; kt = fi >> 4; nt = fi & 15; dst = frags; fragLocal = fi;
  } else if (fi < 128) {
    int f = fi - 64;
    src = W1; stride = 128; kt = f >> 3; nt = f & 7; dst = frags + (size_t)64 * 512; fragLocal = f;
  } else {
    int f = fi - 128;
    src = W2; stride = 384; kt = f / 24; nt = f % 24; dst = frags + (size_t)128 * 512; fragLocal = f;
  }
  int base = (lane & 16) ? 8 : 0;
  int n = nt * 16 + (lane & 15);
  _Float16* out = dst + ((size_t)fragLocal * 32 + lane) * 16;
#pragma unroll
  for (int s = 0; s < 16; ++s) {
    int v = s >> 1, j = s & 1;
    int koff = (v < 4) ? (2 * v + j) : (16 + 2 * (v - 4) + j);
    int k = kt * 32 + base + koff;
    out[s] = (_Float16)src[(size_t)k * stride + n];
  }
}

// ---------------------------------------------------------------------------
// Main fused kernel. 1 wave = 16 batch rows, 4 waves / block.
// ---------------------------------------------------------------------------
#define WAVES 4

__global__ __launch_bounds__(WAVES * 32) void painn_main(
    const float* __restrict__ q, const float* __restrict__ mu,
    const float* __restrict__ dtm, const float* __restrict__ b1,
    const float* __restrict__ b2, const _Float16* __restrict__ frags,
    float* __restrict__ qout, float* __restrict__ muout, int Btot) {
  __shared__ _Float16 act[WAVES][16][264];   // padded: 264 halves/row (no bank conflicts)

  const int lane = threadIdx.x & 31;
  const int w = threadIdx.x >> 5;
  const int b0 = (blockIdx.x * WAVES + w) * 16;
  if (b0 >= Btot) return;

  const int n16 = lane & 15;
  const int hi = lane >> 4;        // 0: rows M=r, 1: rows M=r+8 of C tiles
  const int hi4 = hi * 4;          // K shift (in dwords) for A fragments
  const _Float16* myrow = &act[w][n16][0];

  const _Float16* wmuF = frags;
  const _Float16* w1F = frags + (size_t)64 * 512;
  const _Float16* w2F = frags + (size_t)128 * 512;

  const v8f zero8 = {0.f, 0.f, 0.f, 0.f, 0.f, 0.f, 0.f, 0.f};
  v8f vn[8], vwdot[8];             // sum(mu_V^2), sum(mu_V*mu_W) over components
  v8h muW[3][8];                   // retained mu_W tiles, packed f16 (no spill!)
#pragma unroll
  for (int t = 0; t < 8; ++t) { vn[t] = zero8; vwdot[t] = zero8; }

  // ---------------- GEMM1: mu_mix = mu_i @ W_mu, per component ----------------
  // comp loop FULLY unrolled so muW indices are compile-time constants.
#pragma unroll
  for (int comp = 0; comp < 3; ++comp) {
    __syncthreads();
    // Stage mu_comp tile (16x128 f32 -> f16 LDS)
    for (int t = lane; t < 512; t += 32) {
      int row = t >> 5;
      int c4 = (t & 31) * 4;
      const float4 f = *reinterpret_cast<const float4*>(
          mu + (size_t)(b0 + row) * 384 + comp * 128 + c4);
      _Float16* d = &act[w][row][c4];
      d[0] = (_Float16)f.x; d[1] = (_Float16)f.y;
      d[2] = (_Float16)f.z; d[3] = (_Float16)f.w;
    }
    __syncthreads();

    v16h a[4];
#pragma unroll
    for (int kt = 0; kt < 4; ++kt) a[kt] = load_afrag(myrow, kt, hi4);

#pragma unroll
    for (int ntp = 0; ntp < 8; ++ntp) {
      v8f accV = zero8, accW = zero8;
#pragma unroll
      for (int kt = 0; kt < 4; ++kt) {
        accV = wmma_f16(a[kt], load_bfrag(wmuF, kt * 16 + ntp, lane), accV);
        accW = wmma_f16(a[kt], load_bfrag(wmuF, kt * 16 + 8 + ntp, lane), accW);
      }
      vn[ntp] += accV * accV;
      vwdot[ntp] += accV * accW;
      muW[comp][ntp] = __builtin_convertvector(accW, v8h);
    }
  }

  // Pack the <V,W> dot to f16 now that accumulation is done (frees 32 VGPRs).
  v8h vwdH[8];
#pragma unroll
  for (int t = 0; t < 8; ++t) vwdH[t] = __builtin_convertvector(vwdot[t], v8h);

  // ---------------- Stage ctx = [q, mu_Vn] (16x256 f16) ----------------
  __syncthreads();
  {
    const float4* qs = reinterpret_cast<const float4*>(q + (size_t)b0 * 128);
    for (int t = lane; t < 512; t += 32) {
      float4 f = qs[t];
      int row = t >> 5;
      int c4 = (t & 31) * 4;
      _Float16* d = &act[w][row][c4];
      d[0] = (_Float16)f.x; d[1] = (_Float16)f.y;
      d[2] = (_Float16)f.z; d[3] = (_Float16)f.w;
    }
#pragma unroll
    for (int ntp = 0; ntp < 8; ++ntp)
#pragma unroll
      for (int r = 0; r < 8; ++r)
        act[w][r + 8 * hi][128 + ntp * 16 + n16] =
            (_Float16)__builtin_sqrtf(vn[ntp][r] + 1e-8f);
  }
  __syncthreads();

  // ---------------- GEMM2: h = silu(ctx @ W1 + b1) ----------------
  v16h ca[8];
#pragma unroll
  for (int kt = 0; kt < 8; ++kt) ca[kt] = load_afrag(myrow, kt, hi4);

  // A-frags for GEMM2 are fully in registers; stores below only touch cols 0..127
  // after this wave's in-order DS loads -> safe without extra barrier per nt.
#pragma unroll
  for (int nt = 0; nt < 8; ++nt) {
    v8f acc = zero8;
#pragma unroll
    for (int kt = 0; kt < 8; ++kt)
      acc = wmma_f16(ca[kt], load_bfrag(w1F, kt * 8 + nt, lane), acc);
    float bias = b1[nt * 16 + n16];
#pragma unroll
    for (int r = 0; r < 8; ++r) {
      float v = acc[r] + bias;
      v = v / (1.f + __expf(-v));                  // silu
      act[w][r + 8 * hi][nt * 16 + n16] = (_Float16)v;
    }
  }
  __syncthreads();

  // ---------------- GEMM3: x = h @ W2 + b2, fused epilogue ----------------
  v16h ha[4];
#pragma unroll
  for (int kt = 0; kt < 4; ++kt) ha[kt] = load_afrag(myrow, kt, hi4);

#pragma unroll
  for (int ntp = 0; ntp < 8; ++ntp) {
    v8f xq = zero8, xm = zero8, xqm = zero8;
#pragma unroll
    for (int kt = 0; kt < 4; ++kt) {
      xq  = wmma_f16(ha[kt], load_bfrag(w2F, kt * 24 + ntp, lane), xq);
      xm  = wmma_f16(ha[kt], load_bfrag(w2F, kt * 24 + 8 + ntp, lane), xm);
      xqm = wmma_f16(ha[kt], load_bfrag(w2F, kt * 24 + 16 + ntp, lane), xqm);
    }
    int n = ntp * 16 + n16;                // 0..127
    float bq = b2[n], bm = b2[128 + n], bx = b2[256 + n];
#pragma unroll
    for (int r = 0; r < 8; ++r) {
      int b = b0 + r + 8 * hi;
      float q1 = q[(size_t)b * 128 + n] + xq[r] + bq;        // q + dq_intra
      float q2 = (xqm[r] + bx) * (float)vwdH[ntp][r];        // dqmu_intra * <V,W>
      qout[(size_t)b * 256 + n] = q1;
      qout[(size_t)b * 256 + 128 + n] = q2;
      float dm = xm[r] + bm;                                  // dmu_intra (pre mu_W)
      size_t ob = (size_t)b * 1152;
      size_t db = (size_t)b * 768;
#pragma unroll
      for (int i = 0; i < 3; ++i) {
        float muv = mu[(size_t)b * 384 + i * 128 + n];
        muout[ob + i * 384 + n] = muv + dm * (float)muW[i][ntp][r];  // mu + dmu*mu_W
        muout[ob + i * 384 + 128 + n] = dtm[db + n * 3 + i] * q1;          // tv, f=n
        muout[ob + i * 384 + 256 + n] = dtm[db + (128 + n) * 3 + i] * q2;  // tv, f=128+n
      }
    }
  }
}

// ---------------------------------------------------------------------------
extern "C" void kernel_launch(void* const* d_in, const int* in_sizes, int n_in,
                              void* d_out, int out_size, void* d_ws, size_t ws_size,
                              hipStream_t stream) {
  const float* q   = (const float*)d_in[0];
  const float* mu  = (const float*)d_in[1];
  const float* dtm = (const float*)d_in[2];
  const float* Wmu = (const float*)d_in[3];
  const float* W1  = (const float*)d_in[4];
  const float* b1  = (const float*)d_in[5];
  const float* W2  = (const float*)d_in[6];
  const float* b2  = (const float*)d_in[7];

  const int Btot = in_sizes[0] / 128;          // q is (B,1,128)
  _Float16* frags = (_Float16*)d_ws;           // 224 frags * 1KB = 224KB

  painn_prep<<<224, 32, 0, stream>>>(Wmu, W1, W2, frags);

  float* qout = (float*)d_out;                 // (B,1,256)
  float* muout = qout + (size_t)Btot * 256;    // (B,3,384)
  int blocks = (Btot + (16 * WAVES) - 1) / (16 * WAVES);
  painn_main<<<blocks, WAVES * 32, 0, stream>>>(q, mu, dtm, b1, b2, frags, qout,
                                                muout, Btot);
}
